// Attention_20650202759219
// MI455X (gfx1250) — matrix operand order repaired and checked by reading, not compiled, not run on hardware
//
#include <hip/hip_runtime.h>
#include <hip/hip_bf16.h>
#include <stdint.h>

typedef __attribute__((ext_vector_type(16))) _Float16 v16h;
typedef __attribute__((ext_vector_type(8)))  float    v8f;

#define HEADS 8
#define BB 4
#define CC 512
#define NN 2304   // 48*48 tokens
#define DD 64     // head dim

union Frag {
  v16h v;
  uint32_t u[8];
};

// A-matrix (16x32 f16) lane layout: lane L holds row m=L&15; VGPR j holds K pair:
// base = (j/4)*16 + (L>>4)*8 + (j%4)*2   (CDNA5 ISA 7.12.2)
__device__ __forceinline__ int a_kbase(int j, int half) {
  return ((j >> 2) << 4) + (half << 3) + ((j & 3) << 1);
}
// B-matrix (32x16 f16): lane L holds col n=L&15; VGPR j holds K pair:
// base = (j/4)*16 + (L>>4)*8 + (j%4)*2
__device__ __forceinline__ int b_kbase(int j, int half) {
  return ((j >> 2) << 4) + (half << 3) + ((j & 3) << 1);
}

// 16-lane-group reductions via ds_swizzle (group-of-32 xor mode: offset={xor<<10 | 0x1F})
template <int OFF>
__device__ __forceinline__ float swz(float v) {
  return __int_as_float(__builtin_amdgcn_ds_swizzle(__float_as_int(v), OFF));
}
__device__ __forceinline__ float rmax16(float v) {
  v = fmaxf(v, swz<0x041F>(v));
  v = fmaxf(v, swz<0x081F>(v));
  v = fmaxf(v, swz<0x101F>(v));
  v = fmaxf(v, swz<0x201F>(v));
  return v;
}
__device__ __forceinline__ float rsum16(float v) {
  v += swz<0x041F>(v);
  v += swz<0x081F>(v);
  v += swz<0x101F>(v);
  v += swz<0x201F>(v);
  return v;
}

// CDNA5 async copy: per-lane 16B global -> LDS, tracked by ASYNCcnt.
__device__ __forceinline__ void async_cp16(uint32_t lds_byte_off, const void* g) {
  asm volatile("global_load_async_to_lds_b128 %0, %1, off"
               :: "v"(lds_byte_off), "v"(g) : "memory");
}
__device__ __forceinline__ void wait_async0() {
  asm volatile("s_wait_asynccnt 0x0" ::: "memory");
}
// low 32 bits of a generic pointer into LDS == workgroup LDS byte offset (ISA 10.2)
__device__ __forceinline__ uint32_t lds_off(const void* p) {
  return (uint32_t)(uintptr_t)p;
}

// ---------------------------------------------------------------------------
// Kernel 1: transpose x (b,c,n) f32 -> XT (b,n,c) f16 via LDS tiles
__global__ __launch_bounds__(256) void xpose_kernel(const float* __restrict__ x,
                                                    _Float16* __restrict__ XT) {
  __shared__ float tile[32][33];
  int tt = blockIdx.x * 32;
  int ct = blockIdx.y * 32;
  int b  = blockIdx.z;
  int tx = threadIdx.x & 31;
  int ty = threadIdx.x >> 5;
  const float* xp = x + (size_t)b * CC * NN;
#pragma unroll
  for (int r = ty; r < 32; r += 8)
    tile[r][tx] = xp[(size_t)(ct + r) * NN + tt + tx];
  __syncthreads();
  _Float16* o = XT + (size_t)b * NN * CC;
#pragma unroll
  for (int r = ty; r < 32; r += 8)
    o[(size_t)(tt + r) * CC + ct + tx] = (_Float16)tile[tx][r];
}

// ---------------------------------------------------------------------------
// Kernel 2: convert 4 weight matrices to f16 (packed: q,k,v,sc)
__global__ __launch_bounds__(256) void wconv_kernel(
    const float* __restrict__ Wq, const float* __restrict__ Wk,
    const float* __restrict__ Wv, const float* __restrict__ Wsc,
    _Float16* __restrict__ WF) {
  int i = blockIdx.x * 256 + threadIdx.x;
  int sel = i >> 18;
  int off = i & 262143;
  const float* src = (sel == 0) ? Wq : (sel == 1) ? Wk : (sel == 2) ? Wv : Wsc;
  WF[i] = (_Float16)src[off];
}

// ---------------------------------------------------------------------------
// Kernel 3: fused projections, software-pipelined. One wave = 16x64 tile.
__global__ __launch_bounds__(256) void proj_kernel(
    const _Float16* __restrict__ XT, const _Float16* __restrict__ WF,
    _Float16* __restrict__ Qb, _Float16* __restrict__ Kb,
    _Float16* __restrict__ Vt, float* __restrict__ scout) {
  int lane = threadIdx.x & 31;
  int wave = threadIdx.x >> 5;
  int gw = blockIdx.x * 8 + wave;
  int n64  = gw & 7;   gw >>= 3;
  int mt   = gw % 144; gw /= 144;
  int wsel = gw & 3;
  int b    = gw >> 2;
  int half = lane >> 4, l16 = lane & 15;

  const _Float16* Ap = XT + (size_t)b * NN * CC + (size_t)(mt * 16 + l16) * CC;
  const _Float16* Bp = WF + (size_t)wsel * CC * CC;

  auto load_frags = [&](int k, Frag& a, Frag (&bf)[4]) {
#pragma unroll
    for (int j = 0; j < 8; ++j)
      a.u[j] = *(const uint32_t*)(Ap + k + a_kbase(j, half));
#pragma unroll
    for (int s = 0; s < 4; ++s) {
      int o = n64 * 64 + s * 16 + l16;
#pragma unroll
      for (int j = 0; j < 8; ++j)
        bf[s].u[j] = *(const uint32_t*)(Bp + (size_t)o * CC + k + b_kbase(j, half));
    }
  };

  v8f z = {};
  v8f acc[4] = {z, z, z, z};
  Frag a0, b0[4];
  load_frags(0, a0, b0);
  for (int k = 0; k < CC; k += 32) {
    Frag a1, b1[4];
    int kn = (k + 32 < CC) ? k + 32 : k;   // prefetch next K-step (dup on last)
    load_frags(kn, a1, b1);
#pragma unroll
    for (int s = 0; s < 4; ++s)
      acc[s] = __builtin_amdgcn_wmma_f32_16x16x32_f16(
          false, a0.v, false, b0[s].v, (short)0, acc[s], false, false);
    a0 = a1;
#pragma unroll
    for (int s = 0; s < 4; ++s) b0[s] = b1[s];
  }

#pragma unroll
  for (int s = 0; s < 4; ++s) {
    int o = n64 * 64 + s * 16 + l16;
#pragma unroll
    for (int j = 0; j < 8; ++j) {
      int t = mt * 16 + j + half * 8;
      float val = acc[s][j];
      if (wsel == 3) {
        scout[((size_t)b * CC + o) * NN + t] = val;
      } else if (wsel == 2) {
        Vt[((size_t)b * CC + o) * NN + t] = (_Float16)val;
      } else {
        _Float16* dst = (wsel == 0) ? Qb : Kb;
        int h = o >> 6, d = o & 63;
        dst[(((size_t)b * HEADS + h) * NN + t) * DD + d] = (_Float16)val;
      }
    }
  }
}

// ---------------------------------------------------------------------------
// Kernel 4: flash attention, 8 waves/block over one (b,h). K/V tiles staged in
// LDS with async-to-LDS copies (double buffered, ASYNCcnt), 8x reuse per block.
__global__ __launch_bounds__(256) void flash_kernel(
    const _Float16* __restrict__ Qb, const _Float16* __restrict__ Kb,
    const _Float16* __restrict__ Vt, const float* __restrict__ gamma,
    float* out) {
  __shared__ __align__(16) _Float16 Ksh[2][32 * 64];  // [buf][token][d]
  __shared__ __align__(16) _Float16 Vsh[2][64 * 32];  // [buf][d][token]
  __shared__ __align__(16) _Float16 Pl[8][16 * 36];   // per-wave P transpose

  int tid  = threadIdx.x;
  int lane = tid & 31, wave = tid >> 5;
  int half = lane >> 4, l16 = lane & 15;
  int qt = blockIdx.x * 8 + wave;   // 18 * 8 = 144 q tiles
  int bh = blockIdx.y;
  int b = bh >> 3, h = bh & 7;

  const _Float16* Qp = Qb + ((size_t)bh * NN + qt * 16) * DD;
  const _Float16* Kp = Kb + (size_t)bh * NN * DD;
  const _Float16* Vp = Vt + ((size_t)b * CC + h * DD) * NN;

  // cooperative copy assignment: each thread owns one 16B K chunk + one 16B V chunk
  int kelem = tid * 8;                         // K tile is 2048 f16, contiguous
  int vrow  = tid >> 2, vsub = (tid & 3) * 8;  // V tile: 64 rows x 32 f16
  auto issue_copy = [&](int kt, int buf) {
    async_cp16(lds_off(&Ksh[buf][kelem]), (const void*)(Kp + (size_t)kt * DD + kelem));
    async_cp16(lds_off(&Vsh[buf][vrow * 32 + vsub]),
               (const void*)(Vp + (size_t)vrow * NN + kt + vsub));
  };

  // persistent Q A-fragments (16 x 64)
  Frag qa[2];
#pragma unroll
  for (int dh = 0; dh < 2; ++dh)
#pragma unroll
    for (int j = 0; j < 8; ++j)
      qa[dh].u[j] =
          *(const uint32_t*)(Qp + l16 * DD + dh * 32 + a_kbase(j, half));

  float mrow[8], lrow[8];
#pragma unroll
  for (int j = 0; j < 8; ++j) { mrow[j] = -3.0e38f; lrow[j] = 0.0f; }
  v8f z = {};
  v8f acc[4] = {z, z, z, z};

  issue_copy(0, 0);
  for (int kt = 0, it = 0; kt < NN; kt += 32, ++it) {
    int cur = it & 1;
    wait_async0();        // my copies for tile `it` landed
    __syncthreads();      // everyone's copies landed; prev buffer reads done
    if (kt + 32 < NN) issue_copy(kt + 32, cur ^ 1);

    const _Float16* kb = &Ksh[cur][0];
    const _Float16* vb = &Vsh[cur][0];

    // S tile: 16 q-rows x 32 k-tokens, K-dim = d = 64
    v8f S[2];
#pragma unroll
    for (int ks = 0; ks < 2; ++ks) {
      v8f s = z;
#pragma unroll
      for (int dh = 0; dh < 2; ++dh) {
        Frag bf;
#pragma unroll
        for (int j = 0; j < 8; ++j)
          bf.u[j] = *(const uint32_t*)&kb[(ks * 16 + l16) * 64 + dh * 32 +
                                          b_kbase(j, half)];
        s = __builtin_amdgcn_wmma_f32_16x16x32_f16(false, qa[dh].v, false, bf.v,
                                                   (short)0, s, false, false);
      }
      S[ks] = s;
    }

    // online softmax per row (row = j + 8*half, replicated across 16 lanes)
    float scale[8];
#pragma unroll
    for (int j = 0; j < 8; ++j) {
      float s0 = S[0][j], s1 = S[1][j];
      float mx   = rmax16(fmaxf(s0, s1));
      float mnew = fmaxf(mrow[j], mx);
      float sc = __expf(mrow[j] - mnew);
      float p0 = __expf(s0 - mnew);
      float p1 = __expf(s1 - mnew);
      lrow[j] = lrow[j] * sc + rsum16(p0 + p1);
      mrow[j] = mnew;
      scale[j] = sc;
      int row = j + half * 8;
      Pl[wave][row * 36 + l16]      = (_Float16)p0;
      Pl[wave][row * 36 + 16 + l16] = (_Float16)p1;
    }
#pragma unroll
    for (int dt = 0; dt < 4; ++dt)
#pragma unroll
      for (int j = 0; j < 8; ++j) acc[dt][j] *= scale[j];

    // reload P as A-layout 16x32 fragment (wave-private LDS transpose)
    Frag pa;
#pragma unroll
    for (int j = 0; j < 8; ++j)
      pa.u[j] = *(const uint32_t*)&Pl[wave][l16 * 36 + a_kbase(j, half)];

    // O += P * V
#pragma unroll
    for (int dt = 0; dt < 4; ++dt) {
      Frag bv;
#pragma unroll
      for (int j = 0; j < 8; ++j)
        bv.u[j] = *(const uint32_t*)&vb[(dt * 16 + l16) * 32 + b_kbase(j, half)];
      acc[dt] = __builtin_amdgcn_wmma_f32_16x16x32_f16(
          false, pa.v, false, bv.v, (short)0, acc[dt], false, false);
    }
  }

  // epilogue: out = gamma * (O / l) + out   (out already holds SC projection)
  float* op = out + (size_t)b * CC * NN;
#pragma unroll
  for (int j = 0; j < 8; ++j) {
    float inv = 1.0f / lrow[j];
    int t = qt * 16 + j + half * 8;
#pragma unroll
    for (int dt = 0; dt < 4; ++dt) {
      int c = h * DD + dt * 16 + l16;
      size_t idx = (size_t)c * NN + t;
      op[idx] = fmaf(gamma[c], acc[dt][j] * inv, op[idx]);
    }
  }
}

// ---------------------------------------------------------------------------
extern "C" void kernel_launch(void* const* d_in, const int* in_sizes, int n_in,
                              void* d_out, int out_size, void* d_ws, size_t ws_size,
                              hipStream_t stream) {
  (void)in_sizes; (void)n_in; (void)out_size; (void)ws_size;
  const float* x     = (const float*)d_in[0];
  const float* Wq    = (const float*)d_in[1];
  const float* Wk    = (const float*)d_in[2];
  const float* Wv    = (const float*)d_in[3];
  const float* Wsc   = (const float*)d_in[4];
  const float* gamma = (const float*)d_in[5];
  float* out = (float*)d_out;

  char* ws = (char*)d_ws;
  _Float16* XT = (_Float16*)(ws);                 // B*N*C      = 9,437,184 B
  _Float16* WF = (_Float16*)(ws + 9437184);       // 4*C*C      = 2,097,152 B
  _Float16* Qb = (_Float16*)(ws + 11534336);      // B*H*N*D    = 9,437,184 B
  _Float16* Kb = (_Float16*)(ws + 20971520);      // B*H*N*D    = 9,437,184 B
  _Float16* Vt = (_Float16*)(ws + 30408704);      // B*C*N      = 9,437,184 B

  xpose_kernel<<<dim3(72, 16, 4), 256, 0, stream>>>(x, XT);
  wconv_kernel<<<4096, 256, 0, stream>>>(Wq, Wk, Wv, Wsc, WF);
  proj_kernel<<<2304, 256, 0, stream>>>(XT, WF, Qb, Kb, Vt, out);
  flash_kernel<<<dim3(18, 32), 256, 0, stream>>>(Qb, Kb, Vt, gamma, out);
}
